// Attention_20237885899094
// MI455X (gfx1250) — matrix operand order repaired and checked by reading, not compiled, not run on hardware
//
#include <hip/hip_runtime.h>
#include <math.h>

// ---------------------------------------------------------------------------
// Types / constants
// ---------------------------------------------------------------------------
typedef __attribute__((ext_vector_type(16))) _Float16 v16h;
typedef __attribute__((ext_vector_type(8)))  _Float16 v8h;
typedef __attribute__((ext_vector_type(4)))  _Float16 v4h;
typedef __attribute__((ext_vector_type(8)))  float    v8f;
typedef __attribute__((ext_vector_type(4)))  float    v4f;
typedef __attribute__((ext_vector_type(4)))  int      i4v;
typedef __attribute__((ext_vector_type(2)))  int      i2v;

union HFrag { v16h v; v8h h8[2]; _Float16 h[16]; };

constexpr int Bb = 4;
constexpr int Tt = 2048;
constexpr int Cc = 1024;
constexpr int Hh = 16;
constexpr int Dd = 64;
constexpr int Mm = Bb * Tt;      // 8192 rows

// ---- gfx1250 async (memory -> LDS, ASYNCcnt-tracked, no VGPR data) --------
// Builtin signatures (from hipcc diagnostics): pointees are int4 / int2, with
// global (AS1) source and LDS (AS3) destination.
#if defined(__gfx1250__) && __has_builtin(__builtin_amdgcn_global_load_async_to_lds_b128)
#define HAVE_ASYNC_LDS 1
typedef __attribute__((address_space(1))) i4v gas_i4v;
typedef __attribute__((address_space(3))) i4v las_i4v;
typedef __attribute__((address_space(1))) i2v gas_i2v;
typedef __attribute__((address_space(3))) i2v las_i2v;
__device__ __forceinline__ void async_copy_b128(const _Float16* g, _Float16* l) {
  __builtin_amdgcn_global_load_async_to_lds_b128((gas_i4v*)g, (las_i4v*)l, 0, 0);
}
#if __has_builtin(__builtin_amdgcn_global_load_async_to_lds_b64)
__device__ __forceinline__ void async_copy_b64(const _Float16* g, _Float16* l) {
  __builtin_amdgcn_global_load_async_to_lds_b64((gas_i2v*)g, (las_i2v*)l, 0, 0);
}
#define HAVE_ASYNC_LDS_B64 1
#endif
__device__ __forceinline__ void async_wait0() {
#if __has_builtin(__builtin_amdgcn_s_wait_asynccnt)
  __builtin_amdgcn_s_wait_asynccnt(0);
#else
  asm volatile("s_wait_asynccnt 0" ::: "memory");
#endif
}
#else
#define HAVE_ASYNC_LDS 0
#endif

__device__ __forceinline__ v8f wmma_f16(v16h a, v16h b, v8f c) {
  return __builtin_amdgcn_wmma_f32_16x16x32_f16(
      /*neg_a=*/false, a, /*neg_b=*/false, b,
      /*c_mod=*/(short)0, c, /*reuse_a=*/false, /*reuse_b=*/false);
}

// A fragment (16x32, f16, MxK): lanes 0-15 hold row M=lane with K = k0+0..15;
// lanes 16-31 hold row M=lane-16 with K = k0+16..31.
__device__ __forceinline__ v16h load_a_frag(const _Float16* lds, int pitch,
                                            int row0, int k0, int lane) {
  const int r  = row0 + (lane & 15);
  const int kk = k0 + ((lane >> 4) << 4);
  const _Float16* p = lds + r * pitch + kk;
  HFrag f;
  f.h8[0] = *(const v8h*)(p);
  f.h8[1] = *(const v8h*)(p + 8);
  return f.v;
}

// B fragment (32x16, f16, KxN), source stored as [n][k] row-major (contiguous
// k per n): lanes 0-15 -> column N=n0+lane, K=k0..k0+15; lanes 16-31 ->
// column N=n0+lane-16, K=k0+16..k0+31.
__device__ __forceinline__ v16h load_b_frag(const _Float16* lds, int pitch,
                                            int n0, int k0, int lane) {
  const int n  = n0 + (lane & 15);
  const int kk = k0 + ((lane >> 4) << 4);
  const _Float16* p = lds + n * pitch + kk;
  HFrag f;
  f.h8[0] = *(const v8h*)(p);
  f.h8[1] = *(const v8h*)(p + 8);
  return f.v;
}

// B fragment where source is stored as [k][n] row-major (strided per-k).
__device__ __forceinline__ v16h load_b_frag_strided(const _Float16* lds,
                                                    int pitch, int k0, int n0,
                                                    int lane) {
  const int n  = n0 + (lane & 15);
  const int kk = k0 + ((lane >> 4) << 4);
  HFrag f;
#pragma unroll
  for (int i = 0; i < 16; ++i) f.h[i] = lds[(kk + i) * pitch + n];
  return f.v;
}

// ---------------------------------------------------------------------------
// GEMM: out = X @ W^T + bias
//   MODE 0: X fp32 -> f16 out in [B,H,T,D] layout, scaled (Q, scale=1/8)
//   MODE 1: X fp32 -> f16 out in [B,H,T,D] layout (K, V)
//   MODE 2: X f16  -> fp32 out in [M,C] layout (final projection)
// Tile: 128(M) x 128(N) x 32(K), 256 threads = 8 waves, wave w owns rows
// [w*16, w*16+16) across all 128 columns -> 8 WMMA accumulators per wave.
// ---------------------------------------------------------------------------
constexpr int BM = 128, BN = 128, BK = 32;
constexpr int AP = 40, BP = 40;   // LDS pitches (halves): 16B-aligned rows

template <int MODE>
__global__ __launch_bounds__(256) void proj_kernel(
    const float* __restrict__ Xf, const _Float16* __restrict__ Xh,
    const float* __restrict__ W, const float* __restrict__ bias,
    _Float16* __restrict__ outH, float* __restrict__ outF, float scale) {
  __shared__ __align__(16) _Float16 sA[BM * AP];
  __shared__ __align__(16) _Float16 sB[BN * BP];

  const int tid   = threadIdx.x;
  const int lane  = tid & 31;
  const int wave  = tid >> 5;
  const int nBase = blockIdx.x * BN;
  const int mBase = blockIdx.y * BM;

  v8f acc[8] = {};

  const int sr = tid >> 3;        // staging row 0..31
  const int sc = (tid & 7) * 4;   // staging col 0..28

  for (int k0 = 0; k0 < Cc; k0 += BK) {
    // Stage A tile (X rows, f16)
#pragma unroll
    for (int rr = 0; rr < BM; rr += 32) {
      const int row = sr + rr;
      if (MODE == 2) {
#if HAVE_ASYNC_LDS && defined(HAVE_ASYNC_LDS_B64)
        // raw f16 copy: async memory->LDS (8 bytes), no VGPR staging
        async_copy_b64(Xh + (size_t)(mBase + row) * Cc + k0 + sc,
                       sA + row * AP + sc);
#else
        *(v4h*)(sA + row * AP + sc) =
            *(const v4h*)(Xh + (size_t)(mBase + row) * Cc + k0 + sc);
#endif
      } else {
        v4f fv = *(const v4f*)(Xf + (size_t)(mBase + row) * Cc + k0 + sc);
        v4h hv;
#pragma unroll
        for (int j = 0; j < 4; ++j) hv[j] = (_Float16)fv[j];
        *(v4h*)(sA + row * AP + sc) = hv;
      }
    }
    // Stage B tile (W rows: [n][k], so B[k][n] = W[n][k])
#pragma unroll
    for (int rr = 0; rr < BN; rr += 32) {
      const int row = sr + rr;
      v4f fv = *(const v4f*)(W + (size_t)(nBase + row) * Cc + k0 + sc);
      v4h hv;
#pragma unroll
      for (int j = 0; j < 4; ++j) hv[j] = (_Float16)fv[j];
      *(v4h*)(sB + row * BP + sc) = hv;
    }
#if HAVE_ASYNC_LDS && defined(HAVE_ASYNC_LDS_B64)
    if (MODE == 2) async_wait0();
#endif
    __syncthreads();

    const v16h a = load_a_frag(sA, AP, wave * 16, 0, lane);
#pragma unroll
    for (int nb = 0; nb < 8; ++nb) {
      const v16h bfr = load_b_frag(sB, BP, nb * 16, 0, lane);
      acc[nb] = wmma_f16(a, bfr, acc[nb]);
    }
    __syncthreads();
  }

  // Epilogue.  C/D layout: VGPR i -> row i (lanes 0-15) / row i+8 (lanes
  // 16-31); column = lane & 15.
  const int colLane = lane & 15;
  const int rowSel  = (lane >> 4) * 8;
#pragma unroll
  for (int nb = 0; nb < 8; ++nb) {
    const int n   = nBase + nb * 16 + colLane;
    const float bv = bias[n];
#pragma unroll
    for (int i = 0; i < 8; ++i) {
      const int m = mBase + wave * 16 + rowSel + i;
      float v = acc[nb][i] + bv;
      if (MODE == 2) {
        outF[(size_t)m * Cc + n] = v;
      } else {
        if (MODE == 0) v *= scale;
        const int b = m >> 11, t = m & (Tt - 1);
        const int h = n >> 6,  d = n & (Dd - 1);
        outH[(((size_t)(b * Hh + h)) * Tt + t) * Dd + d] = (_Float16)v;
      }
    }
  }
}

// ---------------------------------------------------------------------------
// Flash attention: block = (b, h, 128-query tile); 8 waves x 16 query rows.
// Q pre-scaled by 1/sqrt(D).  Online softmax over 64-key tiles.
// K/V tiles staged via GLOBAL_LOAD_ASYNC_TO_LDS_B128 (ASYNCcnt) when the
// toolchain exposes the builtin; VGPR staging otherwise.
// ---------------------------------------------------------------------------
constexpr int QT  = 128;   // query rows per block
constexpr int KT  = 64;    // key tile
constexpr int KVP = 72;    // K/V LDS pitch (halves), 16B-aligned rows
constexpr int PP  = 72;    // P LDS pitch

__global__ __launch_bounds__(256) void attn_kernel(
    const _Float16* __restrict__ Q, const _Float16* __restrict__ K,
    const _Float16* __restrict__ V, const int* __restrict__ mask,
    _Float16* __restrict__ Y) {
  __shared__ __align__(16) _Float16 sK[KT * KVP];
  __shared__ __align__(16) _Float16 sV[KT * KVP];
  __shared__ __align__(16) _Float16 sP[8][16 * PP];
  __shared__ float sMaskBias[KT];

  const int tid  = threadIdx.x;
  const int lane = tid & 31;
  const int wave = tid >> 5;
  const int bh   = blockIdx.y;          // 0..63
  const int b    = bh >> 4;
  const int h    = bh & 15;
  const int qBase = blockIdx.x * QT + wave * 16;

  const _Float16* Qhead = Q + (size_t)bh * Tt * Dd;
  const _Float16* Khead = K + (size_t)bh * Tt * Dd;
  const _Float16* Vhead = V + (size_t)bh * Tt * Dd;

  // Load this wave's Q fragments (16 rows x 64 d = two 16x32 A-fragments).
  v16h qf[2];
  {
    const int r = qBase + (lane & 15);
    const _Float16* qp = Qhead + (size_t)r * Dd;
#pragma unroll
    for (int ks = 0; ks < 2; ++ks) {
      const int kk = ks * 32 + ((lane >> 4) << 4);
      HFrag f;
      f.h8[0] = *(const v8h*)(qp + kk);
      f.h8[1] = *(const v8h*)(qp + kk + 8);
      qf[ks] = f.v;
    }
  }

  v8f o[4] = {};
  float mrun[8], lrun[8];
#pragma unroll
  for (int i = 0; i < 8; ++i) { mrun[i] = -1.0e30f; lrun[i] = 0.0f; }

  const int colLane = lane & 15;
  const int rowSel  = (lane >> 4) * 8;

  for (int kb = 0; kb < Tt; kb += KT) {
    // Stage K/V tiles (64 x 64 f16 each) + mask bias.
    {
      const int r = tid >> 2;          // 0..63
      const int c = (tid & 3) * 16;    // 0,16,32,48
      const _Float16* kp = Khead + (size_t)(kb + r) * Dd + c;
      const _Float16* vp = Vhead + (size_t)(kb + r) * Dd + c;
#if HAVE_ASYNC_LDS
      async_copy_b128(kp,     sK + r * KVP + c);
      async_copy_b128(kp + 8, sK + r * KVP + c + 8);
      async_copy_b128(vp,     sV + r * KVP + c);
      async_copy_b128(vp + 8, sV + r * KVP + c + 8);
#else
      *(v8h*)(sK + r * KVP + c)     = *(const v8h*)(kp);
      *(v8h*)(sK + r * KVP + c + 8) = *(const v8h*)(kp + 8);
      *(v8h*)(sV + r * KVP + c)     = *(const v8h*)(vp);
      *(v8h*)(sV + r * KVP + c + 8) = *(const v8h*)(vp + 8);
#endif
      if (kb + KT < Tt) {              // prefetch next tile (global_prefetch_b8)
        __builtin_prefetch(kp + (size_t)KT * Dd, 0, 3);
        __builtin_prefetch(vp + (size_t)KT * Dd, 0, 3);
      }
      if (tid < KT)
        sMaskBias[tid] = (mask[b * Tt + kb + tid] != 0) ? -__builtin_inff() : 0.0f;
    }
#if HAVE_ASYNC_LDS
    async_wait0();                     // drain this wave's ASYNCcnt
#endif
    __syncthreads();

    // S = Q K^T  (B[k=d][n=key] = K[key][d] -> contiguous-k loads from sK)
    v8f s[4] = {};
#pragma unroll
    for (int nb = 0; nb < 4; ++nb) {
#pragma unroll
      for (int ks = 0; ks < 2; ++ks) {
        const v16h bfr = load_b_frag(sK, KVP, nb * 16, ks * 32, lane);
        s[nb] = wmma_f16(qf[ks], bfr, s[nb]);
      }
    }
    // Mask bias (per key column)
#pragma unroll
    for (int nb = 0; nb < 4; ++nb) {
      const float mb = sMaskBias[nb * 16 + colLane];
#pragma unroll
      for (int i = 0; i < 8; ++i) s[nb][i] += mb;
    }
    // Online softmax: row max / exp / row sum (half-wave shuffles).
#pragma unroll
    for (int i = 0; i < 8; ++i) {
      float v = fmaxf(fmaxf(s[0][i], s[1][i]), fmaxf(s[2][i], s[3][i]));
#pragma unroll
      for (int x = 1; x < 16; x <<= 1) v = fmaxf(v, __shfl_xor(v, x, 32));
      const float newm  = fmaxf(mrun[i], v);
      const float alpha = __expf(mrun[i] - newm);
      float rs = 0.0f;
#pragma unroll
      for (int nb = 0; nb < 4; ++nb) {
        const float p = __expf(s[nb][i] - newm);
        s[nb][i] = p;
        rs += p;
      }
#pragma unroll
      for (int x = 1; x < 16; x <<= 1) rs += __shfl_xor(rs, x, 32);
      lrun[i] = lrun[i] * alpha + rs;
      mrun[i] = newm;
#pragma unroll
      for (int j = 0; j < 4; ++j) o[j][i] *= alpha;
    }
    // Spill P (f16) to wave-private LDS in [row][key] layout.
    _Float16* pw = &sP[wave][0];
#pragma unroll
    for (int nb = 0; nb < 4; ++nb)
#pragma unroll
      for (int i = 0; i < 8; ++i)
        pw[(rowSel + i) * PP + nb * 16 + colLane] = (_Float16)s[nb][i];

    // O += P x V   (B[k=key][n=d] = V[key][d] -> strided loads from sV)
#pragma unroll
    for (int ks = 0; ks < 2; ++ks) {
      const v16h pa = load_a_frag(pw, PP, 0, ks * 32, lane);
#pragma unroll
      for (int j = 0; j < 4; ++j) {
        const v16h bv = load_b_frag_strided(sV, KVP, ks * 32, j * 16, lane);
        o[j] = wmma_f16(pa, bv, o[j]);
      }
    }
    __syncthreads();   // before next tile overwrites sK/sV
  }

  // Finalize: divide by row sums, store f16 y1 in [B,T,C] layout.
#pragma unroll
  for (int i = 0; i < 8; ++i) {
    const float inv = 1.0f / lrun[i];
    const int t = qBase + rowSel + i;
#pragma unroll
    for (int j = 0; j < 4; ++j) {
      const size_t idx = ((size_t)b * Tt + t) * Cc + h * Dd + j * 16 + colLane;
      Y[idx] = (_Float16)(o[j][i] * inv);
    }
  }
}

// ---------------------------------------------------------------------------
// Launch
// ---------------------------------------------------------------------------
extern "C" void kernel_launch(void* const* d_in, const int* in_sizes, int n_in,
                              void* d_out, int out_size, void* d_ws,
                              size_t ws_size, hipStream_t stream) {
  (void)in_sizes; (void)n_in; (void)out_size; (void)ws_size;
  const float* x    = (const float*)d_in[0];
  const int*   mask = (const int*)d_in[1];
  const float* Wk   = (const float*)d_in[2];
  const float* bk   = (const float*)d_in[3];
  const float* Wq   = (const float*)d_in[4];
  const float* bq   = (const float*)d_in[5];
  const float* Wv   = (const float*)d_in[6];
  const float* bv   = (const float*)d_in[7];
  const float* Wo   = (const float*)d_in[8];
  const float* bo   = (const float*)d_in[9];
  float* out = (float*)d_out;

  _Float16* ws = (_Float16*)d_ws;
  const size_t sz = (size_t)Mm * Cc;
  _Float16* q  = ws;
  _Float16* k  = ws + sz;
  _Float16* v  = ws + 2 * sz;
  _Float16* y1 = ws + 3 * sz;

  dim3 block(256);
  dim3 gridP(Cc / BN, Mm / BM);       // (8, 64)
  const float scale = 0.125f;         // 1/sqrt(64)

  proj_kernel<0><<<gridP, block, 0, stream>>>(x, nullptr, Wq, bq, q, nullptr, scale);
  proj_kernel<1><<<gridP, block, 0, stream>>>(x, nullptr, Wk, bk, k, nullptr, 1.0f);
  proj_kernel<1><<<gridP, block, 0, stream>>>(x, nullptr, Wv, bv, v, nullptr, 1.0f);

  dim3 gridA(Tt / QT, Bb * Hh);       // (16, 64)
  attn_kernel<<<gridA, block, 0, stream>>>(q, k, v, mask, y1);

  proj_kernel<2><<<gridP, block, 0, stream>>>(nullptr, y1, Wo, bo, nullptr, out, 1.0f);
}
